// NeuralNetwork_48576080117816
// MI455X (gfx1250) — matrix operand order repaired and checked by reading, not compiled, not run on hardware
//
#include <hip/hip_runtime.h>
#include <hip/hip_bf16.h>

typedef __attribute__((ext_vector_type(16))) __bf16 v16bf;
typedef __attribute__((ext_vector_type(8)))  __bf16 v8bf;
typedef __attribute__((ext_vector_type(8)))  float  v8f;
typedef int gv4i __attribute__((vector_size(16)));   // matches builtin's V4i pointee

#define FRAME 64
#define ESN   960
#define COMB  1024
#define LAST  1024
#define TM    128   // batch rows per workgroup (8 waves x 16-row strips)
#define NC    128   // q columns produced per chunk
#define KB    64    // K columns staged per block
#define NKB   (COMB / KB)

#define AS1 __attribute__((address_space(1)))
#define AS3 __attribute__((address_space(3)))

// ---- CDNA5 async global->LDS copy (ASYNCcnt path), builtin with asm fallback ----
__device__ __forceinline__ void async_b128(const void* g, void* l) {
#if __has_builtin(__builtin_amdgcn_global_load_async_to_lds_b128)
  __builtin_amdgcn_global_load_async_to_lds_b128(
      (AS1 gv4i*)(unsigned long long)(uintptr_t)g,
      (AS3 gv4i*)(unsigned int)(uintptr_t)l, 0, 0);
#else
  asm volatile("global_load_async_to_lds_b128 %0, %1, off"
               :
               : "v"((unsigned int)(uintptr_t)l), "v"((unsigned long long)(uintptr_t)g)
               : "memory");
#endif
}

template <int N>
__device__ __forceinline__ void wait_async() {
#if __has_builtin(__builtin_amdgcn_s_wait_asynccnt)
  __builtin_amdgcn_s_wait_asynccnt(N);
#else
  asm volatile("s_wait_asynccnt %0" ::"i"(N) : "memory");
#endif
}

__device__ __forceinline__ v8f wmma_bf16(v16bf a, v16bf b, v8f c) {
  // D = A(16x32 bf16) x B(32x16 bf16) + C(16x16 f32)
  return __builtin_amdgcn_wmma_f32_16x16x32_bf16(false, a, false, b, (short)0, c,
                                                 false, false);
}

// A fragment (16x32 bf16): lane holds its row's K pairs at segments
// [kw+8h, +8) -> v0..3 and [kw+16+8h, +8) -> v4..7.
__device__ __forceinline__ v16bf load_fragA(const __bf16* rowPtr, int kw, int h) {
  v8bf s0 = *(const v8bf*)(rowPtr + kw + 8 * h);
  v8bf s1 = *(const v8bf*)(rowPtr + kw + 16 + 8 * h);
  v16bf r;
#pragma unroll
  for (int i = 0; i < 8; ++i) { r[i] = s0[i]; r[i + 8] = s1[i]; }
  return r;
}

// B fragment (32x16 bf16) from B^T stored [N,K] row-major.
// Lane holds column n = lane&15; K segments [kw+8h, +8) -> v0..3 and [kw+16+8h, +8) -> v4..7.
__device__ __forceinline__ v16bf load_fragB(const __bf16* rowPtr, int kw, int h) {
  v8bf s0 = *(const v8bf*)(rowPtr + kw + 8 * h);
  v8bf s1 = *(const v8bf*)(rowPtr + kw + 16 + 8 * h);
  v16bf r;
#pragma unroll
  for (int i = 0; i < 8; ++i) { r[i] = s0[i]; r[i + 8] = s1[i]; }
  return r;
}

// A fragment from an f32 row, split on the fly into hi/lo bf16 planes.
__device__ __forceinline__ void load_fragA_f32(const float* rowPtr, int kw, int h,
                                               v16bf& ah, v16bf& al) {
#pragma unroll
  for (int i = 0; i < 8; ++i) {
    float x = rowPtr[kw + 8 * h + i];
    __bf16 hx = (__bf16)x;
    ah[i] = hx; al[i] = (__bf16)(x - (float)hx);
    float y = rowPtr[kw + 16 + 8 * h + i];
    __bf16 hy = (__bf16)y;
    ah[i + 8] = hy; al[i + 8] = (__bf16)(y - (float)hy);
  }
}

// ---------------- prep kernels: build hi/lo bf16 weight planes ----------------

// toep[i][j] = params[1023 - i + j];  params[k] = k<1024 ? W1[1023-k][0] : W1[0][k-1023]
__global__ void k_build_toep(const float* __restrict__ W1,
                             __bf16* __restrict__ hi, __bf16* __restrict__ lo) {
  int idx = blockIdx.x * 256 + threadIdx.x;        // 1024*1024 elements, [N=i][K=j]
  int i = idx >> 10, j = idx & 1023;
  int p = 1023 - i + j;                            // 0..2046
  float v = (p < 1024) ? W1[(1023 - p) * COMB] : W1[p - 1023];
  __bf16 h = (__bf16)v;
  hi[idx] = h;
  lo[idx] = (__bf16)(v - (float)h);
}

__global__ void k_split(const float* __restrict__ W,
                        __bf16* __restrict__ hi, __bf16* __restrict__ lo, int n) {
  int idx = blockIdx.x * 256 + threadIdx.x;
  if (idx < n) {
    float v = W[idx];
    __bf16 h = (__bf16)v;
    hi[idx] = h;
    lo[idx] = (__bf16)(v - (float)h);
  }
}

// ---------------- fused main kernel ----------------
//
// Dynamic LDS layout (98304 bytes):
//   sAhi [128][64] bf16 @ 0       (16 KB)
//   sAlo [128][64] bf16 @ 16384   (16 KB)
//   sB buf0: hi @ 32768, lo @ 49152   (32 KB)
//   sB buf1: hi @ 65536, lo @ 81920   (32 KB)
//   sQ  [128][128] f32 @ 0        (64 KB, union with A + buf0, barrier-separated)

__global__ __launch_bounds__(256, 1)
void k_fused(const float* __restrict__ x_frame,   // [B,64]
             const float* __restrict__ h_esn,     // [B,960]
             const __bf16* __restrict__ thi,      // toep hi  [1024,1024] (= B^T of GEMM1)
             const __bf16* __restrict__ tlo,      // toep lo
             const __bf16* __restrict__ wshi,     // W_slope hi [64,1024] (= B^T of GEMM2a)
             const __bf16* __restrict__ wslo,
             const __bf16* __restrict__ wihi,     // W_int hi [64,1024]
             const __bf16* __restrict__ wilo,
             const float* __restrict__ b1,        // [1024]
             const float* __restrict__ b_slope,   // [64]
             const float* __restrict__ b_int,     // [64]
             float* __restrict__ out_slope,       // [B,64]
             float* __restrict__ out_int) {       // [B,64]
  extern __shared__ __attribute__((aligned(128))) unsigned char smem[];
  __bf16* sAhi = (__bf16*)(smem);
  __bf16* sAlo = (__bf16*)(smem + 16384);
  float*  sQ   = (float*)(smem);

  const int tid  = threadIdx.x;
  const int wave = tid >> 5;
  const int lane = tid & 31;
  const int l15  = lane & 15;
  const int hh   = lane >> 4;
  const int m0   = blockIdx.x * TM;

  v8f acc2[8] = {};                          // 0..3 slope tiles, 4..7 intercept tiles

#pragma unroll 1
  for (int nc = 0; nc < LAST / NC; ++nc) {   // 8 chunks of 128 q-columns
    const int n0 = nc * NC;
    v8f C[8] = {};

    __syncthreads();                         // phase-2 readers of sQ done (aliases buf0/A)

    // preload B tile for kb=0 into buffer 0 (async, no VGPR staging)
    {
      __bf16* sbh = (__bf16*)(smem + 32768);
      __bf16* sbl = (__bf16*)(smem + 49152);
#pragma unroll
      for (int i = 0; i < 4; ++i) {
        int f = (i * 256 + tid) * 8;         // 8 bf16 = 16 bytes per op
        int n = n0 + (f >> 6), k = (f & 63);
        async_b128(thi + n * COMB + k, sbh + f);
        async_b128(tlo + n * COMB + k, sbl + f);
      }
    }

    // ---- phase 1: q[:,n0:n0+128] = combined @ toep^T  (K = 1024) ----
#pragma unroll 1
    for (int kb = 0; kb < NKB; ++kb) {
      const int cur = kb & 1;
      __bf16* sBhi = (__bf16*)(smem + 32768 + cur * 32768);
      __bf16* sBlo = (__bf16*)(smem + 49152 + cur * 32768);

      __syncthreads();                       // everyone done computing kb-1

      if (kb + 1 < NKB) {                    // async-prefetch next B tile
        __bf16* nbh = (__bf16*)(smem + 32768 + ((kb + 1) & 1) * 32768);
        __bf16* nbl = (__bf16*)(smem + 49152 + ((kb + 1) & 1) * 32768);
#pragma unroll
        for (int i = 0; i < 4; ++i) {
          int f = (i * 256 + tid) * 8;
          int n = n0 + (f >> 6), k = (kb + 1) * KB + (f & 63);
          async_b128(thi + n * COMB + k, nbh + f);
          async_b128(tlo + n * COMB + k, nbl + f);
        }
      }

      // stage A: 128 rows x 64 K of combined, converted to hi/lo bf16
      // (overlaps with in-flight async B copies)
#pragma unroll
      for (int i = 0; i < 32; ++i) {
        int e = i * 256 + tid;               // 8192 elems
        int r = e >> 6, c = e & 63;
        int b = m0 + r;
        float v = (kb == 0) ? x_frame[b * FRAME + c]
                            : h_esn[b * ESN + (kb - 1) * 64 + c];
        __bf16 vh = (__bf16)v;
        sAhi[e] = vh;
        sAlo[e] = (__bf16)(v - (float)vh);
      }

      if (kb + 1 < NKB) wait_async<8>();     // current tile landed; next still in flight
      else              wait_async<0>();
      __syncthreads();                       // publish A + B(kb)

#pragma unroll
      for (int ks = 0; ks < KB / 32; ++ks) {
        const int kw = ks * 32;
        const __bf16* arow = sAhi + (wave * 16 + l15) * KB;
        const __bf16* arol = sAlo + (wave * 16 + l15) * KB;
        v16bf Ah = load_fragA(arow, kw, hh);
        v16bf Al = load_fragA(arol, kw, hh);
#pragma unroll
        for (int t = 0; t < 8; ++t) {
          const __bf16* brow = sBhi + (t * 16 + l15) * KB;
          const __bf16* brol = sBlo + (t * 16 + l15) * KB;
          v16bf Bh = load_fragB(brow, kw, hh);
          v16bf Bl = load_fragB(brol, kw, hh);
          C[t] = wmma_bf16(Ah, Bh, C[t]);
          C[t] = wmma_bf16(Ah, Bl, C[t]);
          C[t] = wmma_bf16(Al, Bh, C[t]);
        }
      }
    }

    // ---- bias + ReLU, spill q chunk (f32) to LDS ----
    __syncthreads();
#pragma unroll
    for (int t = 0; t < 8; ++t) {
      float bv = b1[n0 + t * 16 + l15];
#pragma unroll
      for (int r = 0; r < 8; ++r) {
        float q = C[t][r] + bv;
        q = fmaxf(q, 0.0f);
        int row = wave * 16 + 8 * hh + r;    // C/D layout: M = r + 8h, N = lane&15
        sQ[row * NC + t * 16 + l15] = q;
      }
    }
    __syncthreads();

    // ---- phase 2: accumulate slope/int partials over this 128-wide K2 chunk ----
#pragma unroll
    for (int ks2 = 0; ks2 < NC / 32; ++ks2) {
      const int kw = ks2 * 32;
      const float* qrow = sQ + (wave * 16 + l15) * NC;
      v16bf Qh, Ql;
      load_fragA_f32(qrow, kw, hh, Qh, Ql);
#pragma unroll
      for (int t = 0; t < 4; ++t) {          // slope
        const __bf16* rh = wshi + (t * 16 + l15) * LAST + n0;
        const __bf16* rl = wslo + (t * 16 + l15) * LAST + n0;
        v16bf Bh = load_fragB(rh, kw, hh);
        v16bf Bl = load_fragB(rl, kw, hh);
        acc2[t] = wmma_bf16(Qh, Bh, acc2[t]);
        acc2[t] = wmma_bf16(Qh, Bl, acc2[t]);
        acc2[t] = wmma_bf16(Ql, Bh, acc2[t]);
      }
#pragma unroll
      for (int t = 0; t < 4; ++t) {          // intercept
        const __bf16* rh = wihi + (t * 16 + l15) * LAST + n0;
        const __bf16* rl = wilo + (t * 16 + l15) * LAST + n0;
        v16bf Bh = load_fragB(rh, kw, hh);
        v16bf Bl = load_fragB(rl, kw, hh);
        acc2[4 + t] = wmma_bf16(Qh, Bh, acc2[4 + t]);
        acc2[4 + t] = wmma_bf16(Qh, Bl, acc2[4 + t]);
        acc2[4 + t] = wmma_bf16(Ql, Bh, acc2[4 + t]);
      }
    }
  }

  // ---- epilogue: biases, tanh, store both outputs ----
#pragma unroll
  for (int t = 0; t < 4; ++t) {
    int n2 = t * 16 + l15;
    float bs = b_slope[n2];
    float bi = b_int[n2];
#pragma unroll
    for (int r = 0; r < 8; ++r) {
      int row = m0 + wave * 16 + 8 * hh + r;
      out_slope[row * FRAME + n2] = tanhf(acc2[t][r] + bs);
      out_int[row * FRAME + n2]   = acc2[4 + t][r] + bi;
    }
  }
}

// ---------------- launch ----------------

extern "C" void kernel_launch(void* const* d_in, const int* in_sizes, int n_in,
                              void* d_out, int out_size, void* d_ws, size_t ws_size,
                              hipStream_t stream) {
  const float* x_frame = (const float*)d_in[0];
  const float* h_esn   = (const float*)d_in[1];
  const float* W1      = (const float*)d_in[2];
  const float* b1      = (const float*)d_in[3];
  const float* W_slope = (const float*)d_in[4];
  const float* b_slope = (const float*)d_in[5];
  const float* W_int   = (const float*)d_in[6];
  const float* b_int   = (const float*)d_in[7];

  const int B = in_sizes[0] / FRAME;         // 65536

  __bf16* thi  = (__bf16*)d_ws;                  // 1024*1024
  __bf16* tlo  = thi + (size_t)LAST * COMB;      // 1024*1024
  __bf16* wshi = tlo + (size_t)LAST * COMB;      // 64*1024
  __bf16* wslo = wshi + (size_t)FRAME * LAST;
  __bf16* wihi = wslo + (size_t)FRAME * LAST;
  __bf16* wilo = wihi + (size_t)FRAME * LAST;

  k_build_toep<<<(LAST * COMB) / 256, 256, 0, stream>>>(W1, thi, tlo);
  k_split<<<(FRAME * LAST) / 256, 256, 0, stream>>>(W_slope, wshi, wslo, FRAME * LAST);
  k_split<<<(FRAME * LAST) / 256, 256, 0, stream>>>(W_int, wihi, wilo, FRAME * LAST);

  float* out_slope = (float*)d_out;
  float* out_int   = out_slope + (size_t)B * FRAME;

  k_fused<<<B / TM, 256, 98304, stream>>>(x_frame, h_esn, thi, tlo, wshi, wslo,
                                          wihi, wilo, b1, b_slope, b_int,
                                          out_slope, out_int);
}
